// Attention_37228776522370
// MI455X (gfx1250) — matrix operand order repaired and checked by reading, not compiled, not run on hardware
//
#include <hip/hip_runtime.h>
#include <hip/hip_bf16.h>
#include <cstdint>
#include <cstddef>

// Problem constants (B=2, N=2048, D=1024, H=16, SUB=64)
#define BB 2
#define NN 2048
#define DD 1024
#define HH 16
#define SUBD 64
#define SCALE_F 0.125f  // 64^-0.5

typedef __attribute__((ext_vector_type(16))) __bf16 v16bf;
typedef __attribute__((ext_vector_type(8)))  float  v8f;
typedef __attribute__((ext_vector_type(4)))  unsigned int v4u;
typedef __attribute__((ext_vector_type(8)))  int v8i;
typedef __attribute__((ext_vector_type(4)))  int v4i;

union Frag {
  v16bf bf;
  unsigned short us[16];
  uint4 q[2];
};

__device__ __forceinline__ unsigned short f2bf(float f) {
  unsigned int u = __float_as_uint(f);
  u += 0x7FFFu + ((u >> 16) & 1u);     // round-to-nearest-even
  return (unsigned short)(u >> 16);
}

__device__ __forceinline__ v8f wmma_bf16(v16bf a, v16bf b, v8f c) {
  return __builtin_amdgcn_wmma_f32_16x16x32_bf16(false, a, false, b, (short)0, c, false, false);
}

__device__ __forceinline__ v16bf load_frag2(const unsigned short* p0, const unsigned short* p1) {
  Frag f;
  f.q[0] = *(const uint4*)p0;
  f.q[1] = *(const uint4*)p1;
  return f.bf;
}
__device__ __forceinline__ v16bf load_frag1(const unsigned short* p) {  // 16 contiguous bf16
  return load_frag2(p, p + 8);
}
__device__ __forceinline__ v16bf load_frag_f32(const float* p0, const float* p1) {
  Frag f;
#pragma unroll
  for (int i = 0; i < 8; ++i) f.us[i] = f2bf(p0[i]);
#pragma unroll
  for (int i = 0; i < 8; ++i) f.us[8 + i] = f2bf(p1[i]);
  return f.bf;
}

// ---- Tensor Data Mover: DMA a 2D bf16 tile (tile_w x tile_h, row stride in elems) to LDS.
// D# Group0/1 bitfields per CDNA5 ISA. 6-arg builtin form
// (this toolchain): (uint32x4 g0, int32x8 g1, int32x4 g2, int32x4 g3, int32x8 g4, i32 cpol).
__device__ __forceinline__ void tdm_load_2d(unsigned lds_off, const void* gptr,
                                            unsigned tile_w, unsigned tile_h,
                                            unsigned row_stride_elems,
                                            unsigned tensor_w, unsigned tensor_h) {
  unsigned long long ga = (unsigned long long)(uintptr_t)gptr;
  v4u g0;
  g0[0] = 1u;                                   // count=1, user descriptor
  g0[1] = lds_off;                              // lds_addr (bytes)
  g0[2] = (unsigned)(ga & 0xFFFFFFFFu);         // global_addr[31:0]
  g0[3] = (unsigned)((ga >> 32) & 0x01FFFFFFu)  // global_addr[56:32]
        | (2u << 30);                           // type=2 ("image")
  v8i g1;
  g1[0] = 0x00010000;                           // data_size=1 (2 bytes); no multicast/pad/iter
  g1[1] = (int)((tensor_w & 0xFFFFu) << 16);    // tensor_dim0[15:0]
  g1[2] = (int)((tensor_w >> 16) | ((tensor_h & 0xFFFFu) << 16));   // td0[31:16] | td1[15:0]
  g1[3] = (int)((tensor_h >> 16) | (tile_w << 16));                 // td1[31:16] | tile_dim0
  g1[4] = (int)tile_h;                          // tile_dim1; tile_dim2=0
  g1[5] = (int)row_stride_elems;                // tensor_dim0_stride[31:0]
  g1[6] = 0;                                    // stride[47:32]=0, dim1_stride[15:0]=0
  g1[7] = 0;
  v4i g2 = {0, 0, 0, 0};
  v4i g3 = {0, 0, 0, 0};
  v8i g4 = {0, 0, 0, 0, 0, 0, 0, 0};            // unused extra group (2D tensor)
  __builtin_amdgcn_tensor_load_to_lds(g0, g1, g2, g3, g4, 0);
}

// ---------------------------------------------------------------- fp32 -> bf16 (x)
__global__ void cvt_f32_bf16(const float* __restrict__ src, unsigned short* __restrict__ dst, int n4) {
  int i = blockIdx.x * blockDim.x + threadIdx.x;
  if (i >= n4) return;
  float4 f = ((const float4*)src)[i];
  uint2 u;
  u.x = (unsigned)f2bf(f.x) | ((unsigned)f2bf(f.y) << 16);
  u.y = (unsigned)f2bf(f.z) | ((unsigned)f2bf(f.w) << 16);
  ((uint2*)dst)[i] = u;
}

// ------------------------------------------- weights: fp32 [K][N] -> bf16 [N][K] (LDS transpose)
__global__ __launch_bounds__(256) void transpose_cvt(const float* __restrict__ src,
                                                     unsigned short* __restrict__ dst,
                                                     int rows /*K*/, int cols /*N*/) {
  __shared__ float t[32][33];
  int c0 = blockIdx.x * 32, r0 = blockIdx.y * 32;
  for (int j = threadIdx.y; j < 32; j += 8)
    t[j][threadIdx.x] = src[(size_t)(r0 + j) * cols + c0 + threadIdx.x];
  __syncthreads();
  for (int j = threadIdx.y; j < 32; j += 8)
    dst[(size_t)(c0 + j) * rows + r0 + threadIdx.x] = f2bf(t[threadIdx.x][j]);
}

// --------------------------------------------------- QKV GEMM: X(4096x1024) * WqkvT(3072x1024)^T
// 128x64 tile / 256-thread WG. A and B tiles DMA'd by the TDM (wave 0), double-buffered.
// Epilogue scatters into Q[b][h][n][sub], K[b][h][n][sub], Vt[b][h][sub][n] (bf16).
__global__ __launch_bounds__(256) void qkv_gemm(
    const unsigned short* __restrict__ Xb, const unsigned short* __restrict__ WT,
    const float* __restrict__ bias,
    unsigned short* __restrict__ Qg, unsigned short* __restrict__ Kg, unsigned short* __restrict__ Vt) {
  __shared__ unsigned short As[2][128 * 32];
  __shared__ unsigned short Bs[2][64 * 32];
  const int lane = threadIdx.x & 31, wave = threadIdx.x >> 5, hb = (threadIdx.x & 31) >> 4;
  const int mbase = blockIdx.y * 128, nbase = blockIdx.x * 64;
  const unsigned ldsA[2] = {(unsigned)(uintptr_t)&As[0][0], (unsigned)(uintptr_t)&As[1][0]};
  const unsigned ldsB[2] = {(unsigned)(uintptr_t)&Bs[0][0], (unsigned)(uintptr_t)&Bs[1][0]};
  v8f acc[4] = {};
  if (wave == 0) {
    tdm_load_2d(ldsA[0], Xb + (size_t)mbase * DD, 32, 128, DD, DD, 1u << 20);
    tdm_load_2d(ldsB[0], WT + (size_t)nbase * DD, 32, 64, DD, DD, 1u << 20);
  }
  int buf = 0;
  for (int k0 = 0; k0 < DD; k0 += 32, buf ^= 1) {
    if (wave == 0) {
      if (k0 + 32 < DD) {
        tdm_load_2d(ldsA[buf ^ 1], Xb + (size_t)mbase * DD + k0 + 32, 32, 128, DD, DD, 1u << 20);
        tdm_load_2d(ldsB[buf ^ 1], WT + (size_t)nbase * DD + k0 + 32, 32, 64, DD, DD, 1u << 20);
        __builtin_amdgcn_s_wait_tensorcnt(2);   // current buffer's 2 DMAs retired
      } else {
        __builtin_amdgcn_s_wait_tensorcnt(0);
      }
    }
    __syncthreads();
    const unsigned short* arow = &As[buf][(wave * 16 + (lane & 15)) * 32];
    v16bf af = load_frag2(arow + 8 * hb, arow + 16 + 8 * hb);
#pragma unroll
    for (int nt = 0; nt < 4; ++nt) {
      v16bf bfrag = load_frag2(&Bs[buf][(nt * 16 + (lane & 15)) * 32 + 8 * hb], &Bs[buf][(nt * 16 + (lane & 15)) * 32 + 16 + 8 * hb]);
      acc[nt] = wmma_bf16(af, bfrag, acc[nt]);
    }
    __syncthreads();
  }
  // epilogue: + bias, scatter to Q / K / Vt (pointer+stride hoisted out of the store loop)
  const int b = mbase >> 11;
  const int n0 = (mbase & (NN - 1)) + wave * 16 + 8 * hb;
#pragma unroll
  for (int nt = 0; nt < 4; ++nt) {
    int c = nbase + nt * 16 + (lane & 15);            // column in [0, 3072)
    int three = c >> 10, h = (c >> 6) & 15, sub = c & 63;
    float bv = bias[c];
    size_t bh = (size_t)(b * HH + h);
    unsigned short* dst;
    size_t base;
    size_t stride;
    if (three == 0)      { dst = Qg; base = (bh * NN + n0) * SUBD + sub; stride = SUBD; }
    else if (three == 1) { dst = Kg; base = (bh * NN + n0) * SUBD + sub; stride = SUBD; }
    else                 { dst = Vt; base = (bh * SUBD + sub) * NN + n0; stride = 1; }
#pragma unroll
    for (int v = 0; v < 8; ++v)
      dst[base + (size_t)v * stride] = f2bf(acc[nt][v] + bv);
  }
}

// --------------------------------------------------- Attention (one WG per (b,h,16-query block))
__global__ __launch_bounds__(128) void attn_kernel(
    const unsigned short* __restrict__ Qg, const unsigned short* __restrict__ Kg,
    const unsigned short* __restrict__ Vt, unsigned short* __restrict__ Og) {
  extern __shared__ float smem[];
  float* S = smem;                     // 16 x 2048 fp32 scores
  float* red    = smem + 16 * NN;      // 128 partial maxima
  float* red2   = red + 128;           // 128 partial sums
  float* rowinv = red2 + 128;          // 16
  const int lane = threadIdx.x & 31, wave = threadIdx.x >> 5, hb = (threadIdx.x & 31) >> 4;
  const int bh = blockIdx.y;           // b*16 + h
  const int qbase = blockIdx.x * 16;

  // -------- phase 1: S = (Q K^T) * scale, wave w covers keys [w*512, w*512+512)
  const unsigned short* qrow = Qg + ((size_t)bh * NN + qbase + (lane & 15)) * SUBD;
  v16bf q0 = load_frag2(qrow + 8 * hb,      qrow + 16 + 8 * hb);   // subs 0..31
  v16bf q1 = load_frag2(qrow + 32 + 8 * hb, qrow + 48 + 8 * hb);   // subs 32..63
  for (int kt = wave * 32; kt < wave * 32 + 32; ++kt) {
    int key = kt * 16 + (lane & 15);
    const unsigned short* krow = Kg + ((size_t)bh * NN + key) * SUBD;
    v8f acc = {};
    acc = wmma_bf16(q0, load_frag2(krow + 8 * hb, krow + 16 + 8 * hb), acc);
    acc = wmma_bf16(q1, load_frag2(krow + 32 + 8 * hb, krow + 48 + 8 * hb), acc);
#pragma unroll
    for (int v = 0; v < 8; ++v)
      S[(v + 8 * hb) * NN + key] = acc[v] * SCALE_F;
  }
  __syncthreads();

  // -------- phase 2: softmax per row (8 threads x 256 cols per row); unnormalized exp
  {
    int r = threadIdx.x >> 3, seg = threadIdx.x & 7;
    float* row = S + r * NN + seg * 256;
    float m = -3.0e38f;
    for (int i = 0; i < 256; ++i) m = fmaxf(m, row[i]);
    red[r * 8 + seg] = m;
    __syncthreads();
    float rm = red[r * 8];
#pragma unroll
    for (int i = 1; i < 8; ++i) rm = fmaxf(rm, red[r * 8 + i]);
    float s = 0.f;
    for (int i = 0; i < 256; ++i) { float e = __expf(row[i] - rm); row[i] = e; s += e; }
    red2[r * 8 + seg] = s;
    __syncthreads();
    if (seg == 0) {
      float t = 0.f;
#pragma unroll
      for (int i = 0; i < 8; ++i) t += red2[r * 8 + i];
      rowinv[r] = 1.0f / t;
    }
  }
  __syncthreads();

  // -------- phase 3: O = P V, wave w -> sub columns [w*16, w*16+16)
  {
    int sub = wave * 16 + (lane & 15);
    const unsigned short* vrow = Vt + ((size_t)bh * SUBD + sub) * NN;   // contiguous in key
    const float* prow = S + (lane & 15) * NN;                           // A row = lane%16
    v8f acc = {};
    for (int kb = 0; kb < NN; kb += 32) {
      v16bf pf = load_frag_f32(prow + kb + 8 * hb, prow + kb + 16 + 8 * hb);
      v16bf vf = load_frag2(vrow + kb + 8 * hb, vrow + kb + 16 + 8 * hb);
      acc = wmma_bf16(pf, vf, acc);
    }
    int b = bh >> 4, h = bh & 15;
#pragma unroll
    for (int v = 0; v < 8; ++v) {
      int M = v + 8 * hb;
      float o = acc[v] * rowinv[M];
      Og[((size_t)(b * NN + qbase + M)) * DD + h * SUBD + sub] = f2bf(o);
    }
  }
}

// --------------------------------------------------- Output GEMM: Ao(4096x1024) * WoT(1024x1024)^T
__global__ __launch_bounds__(256) void out_gemm(
    const unsigned short* __restrict__ Ab, const unsigned short* __restrict__ WT,
    const float* __restrict__ bias, float* __restrict__ out) {
  __shared__ unsigned short As[2][128 * 32];
  __shared__ unsigned short Bs[2][64 * 32];
  const int lane = threadIdx.x & 31, wave = threadIdx.x >> 5, hb = (threadIdx.x & 31) >> 4;
  const int mbase = blockIdx.y * 128, nbase = blockIdx.x * 64;
  const unsigned ldsA[2] = {(unsigned)(uintptr_t)&As[0][0], (unsigned)(uintptr_t)&As[1][0]};
  const unsigned ldsB[2] = {(unsigned)(uintptr_t)&Bs[0][0], (unsigned)(uintptr_t)&Bs[1][0]};
  v8f acc[4] = {};
  if (wave == 0) {
    tdm_load_2d(ldsA[0], Ab + (size_t)mbase * DD, 32, 128, DD, DD, 1u << 20);
    tdm_load_2d(ldsB[0], WT + (size_t)nbase * DD, 32, 64, DD, DD, 1u << 20);
  }
  int buf = 0;
  for (int k0 = 0; k0 < DD; k0 += 32, buf ^= 1) {
    if (wave == 0) {
      if (k0 + 32 < DD) {
        tdm_load_2d(ldsA[buf ^ 1], Ab + (size_t)mbase * DD + k0 + 32, 32, 128, DD, DD, 1u << 20);
        tdm_load_2d(ldsB[buf ^ 1], WT + (size_t)nbase * DD + k0 + 32, 32, 64, DD, DD, 1u << 20);
        __builtin_amdgcn_s_wait_tensorcnt(2);
      } else {
        __builtin_amdgcn_s_wait_tensorcnt(0);
      }
    }
    __syncthreads();
    const unsigned short* arow = &As[buf][(wave * 16 + (lane & 15)) * 32];
    v16bf af = load_frag2(arow + 8 * hb, arow + 16 + 8 * hb);
#pragma unroll
    for (int nt = 0; nt < 4; ++nt) {
      v16bf bfrag = load_frag2(&Bs[buf][(nt * 16 + (lane & 15)) * 32 + 8 * hb], &Bs[buf][(nt * 16 + (lane & 15)) * 32 + 16 + 8 * hb]);
      acc[nt] = wmma_bf16(af, bfrag, acc[nt]);
    }
    __syncthreads();
  }
#pragma unroll
  for (int nt = 0; nt < 4; ++nt) {
    int c = nbase + nt * 16 + (lane & 15);
    float bv = bias[c];
#pragma unroll
    for (int v = 0; v < 8; ++v) {
      int row = mbase + wave * 16 + v + 8 * hb;
      out[(size_t)row * DD + c] = acc[nt][v] + bv;
    }
  }
}

extern "C" void kernel_launch(void* const* d_in, const int* in_sizes, int n_in,
                              void* d_out, int out_size, void* d_ws, size_t ws_size,
                              hipStream_t stream) {
  (void)in_sizes; (void)n_in; (void)out_size; (void)ws_size;
  const float* x    = (const float*)d_in[0];
  const float* Wqkv = (const float*)d_in[1];
  const float* bqkv = (const float*)d_in[2];
  const float* Wo   = (const float*)d_in[3];
  const float* bo   = (const float*)d_in[4];
  float* out = (float*)d_out;

  // workspace carve-up (bf16 buffers), 48 MB total
  char* w = (char*)d_ws;
  unsigned short* xbf   = (unsigned short*)w; w += (size_t)BB * NN * DD * 2;        // 8 MB
  unsigned short* wqkvT = (unsigned short*)w; w += (size_t)3 * DD * DD * 2;         // 6 MB  [3072][1024]
  unsigned short* woT   = (unsigned short*)w; w += (size_t)DD * DD * 2;             // 2 MB  [1024][1024]
  unsigned short* Qg    = (unsigned short*)w; w += (size_t)BB * HH * NN * SUBD * 2; // 8 MB
  unsigned short* Kg    = (unsigned short*)w; w += (size_t)BB * HH * NN * SUBD * 2; // 8 MB
  unsigned short* Vt    = (unsigned short*)w; w += (size_t)BB * HH * NN * SUBD * 2; // 8 MB
  unsigned short* Ao    = (unsigned short*)w; w += (size_t)BB * NN * DD * 2;        // 8 MB

  cvt_f32_bf16<<<(BB * NN * DD / 4) / 256, 256, 0, stream>>>(x, xbf, BB * NN * DD / 4);
  transpose_cvt<<<dim3(3 * DD / 32, DD / 32), dim3(32, 8), 0, stream>>>(Wqkv, wqkvT, DD, 3 * DD);
  transpose_cvt<<<dim3(DD / 32, DD / 32), dim3(32, 8), 0, stream>>>(Wo, woT, DD, DD);

  qkv_gemm<<<dim3(48, 32), 256, 0, stream>>>(xbf, wqkvT, bqkv, Qg, Kg, Vt);

  const int smem_bytes = (16 * NN + 128 + 128 + 16) * (int)sizeof(float);  // ~129 KB
  (void)hipFuncSetAttribute(reinterpret_cast<const void*>(attn_kernel),
                            hipFuncAttributeMaxDynamicSharedMemorySize, smem_bytes);
  attn_kernel<<<dim3(NN / 16, BB * HH), 128, smem_bytes, stream>>>(Qg, Kg, Vt, Ao);

  out_gemm<<<dim3(DD / 64, BB * NN / 128), 256, 0, stream>>>(Ao, woT, bo, out);
}
